// CausalSelfAttention_49718541418549
// MI455X (gfx1250) — matrix operand order repaired and checked by reading, not compiled, not run on hardware
//
#include <hip/hip_runtime.h>

typedef float    v8f  __attribute__((ext_vector_type(8)));
typedef _Float16 v16h __attribute__((ext_vector_type(16)));
typedef _Float16 v8h  __attribute__((ext_vector_type(8)));

static constexpr int kB  = 2;
static constexpr int kT  = 2048;
static constexpr int kD  = 1024;
static constexpr int kH  = 16;
static constexpr int kHD = 64;
static constexpr int kM  = kB * kT;   // 4096 total rows
static constexpr int kNQ = 3 * kD;    // 3072 qkv columns

__device__ __forceinline__ v16h cat8(v8h lo, v8h hi) {
  return __builtin_shufflevector(lo, hi, 0,1,2,3,4,5,6,7,8,9,10,11,12,13,14,15);
}

__device__ __forceinline__ v8f wmma_f16(v16h a, v16h b, v8f c) {
  return __builtin_amdgcn_wmma_f32_16x16x32_f16(false, a, false, b, (short)0, c,
                                                false, false);
}

// reductions across the 16 lanes of a half-wave (masks 1,2,4,8 stay in-half)
__device__ __forceinline__ float redmax16(float v) {
  v = fmaxf(v, __shfl_xor(v, 1, 32));
  v = fmaxf(v, __shfl_xor(v, 2, 32));
  v = fmaxf(v, __shfl_xor(v, 4, 32));
  v = fmaxf(v, __shfl_xor(v, 8, 32));
  return v;
}
__device__ __forceinline__ float redsum16(float v) {
  v += __shfl_xor(v, 1, 32);
  v += __shfl_xor(v, 2, 32);
  v += __shfl_xor(v, 4, 32);
  v += __shfl_xor(v, 8, 32);
  return v;
}

__global__ void cvt_f32_to_f16(const float* __restrict__ s,
                               _Float16* __restrict__ d, int n) {
  int i = blockIdx.x * blockDim.x + threadIdx.x;
  int stride = gridDim.x * blockDim.x;
  for (; i < n; i += stride) d[i] = (_Float16)s[i];
}

// --------------------------------------------------------------------------
// qkv = x @ w_qkv^T ; epilogue scatters into per-head q/k/v [B,H,T,64] (f16)
// One wave computes a 16(M) x 64(N) tile. Single-buffered K loop (chunk 32):
// spill-free; latency hidden by wave-level parallelism (12288 waves).
// --------------------------------------------------------------------------
__global__ void qkv_gemm(const _Float16* __restrict__ xh,
                         const _Float16* __restrict__ wh,
                         _Float16* __restrict__ qh,
                         _Float16* __restrict__ kh,
                         _Float16* __restrict__ vh) {
  const int lane = threadIdx.x & 31;
  const int wv   = threadIdx.x >> 5;
  const int hf   = lane >> 4;
  const int lc   = lane & 15;
  const int gw   = blockIdx.x * 4 + wv;
  const int tilesN = kNQ / 64;  // 48
  const int m0 = (gw / tilesN) * 16;
  const int n0 = (gw % tilesN) * 64;

  v8f acc[4] = {};
  const _Float16* ap = xh + (size_t)(m0 + lc) * kD + hf * 16;
  for (int kt = 0; kt < kD; kt += 32) {
    v16h A = cat8(*(const v8h*)(ap + kt), *(const v8h*)(ap + kt + 8));
    __builtin_prefetch(ap + kt + 64, 0, 3);
#pragma unroll
    for (int c = 0; c < 4; ++c) {
      const _Float16* bp = wh + (size_t)(n0 + c * 16 + lc) * kD + kt + hf * 16;
      v16h Bf = cat8(*(const v8h*)bp, *(const v8h*)(bp + 8));
      acc[c] = wmma_f16(A, Bf, acc[c]);
    }
  }
#pragma unroll
  for (int c = 0; c < 4; ++c) {
#pragma unroll
    for (int r = 0; r < 8; ++r) {
      int m = m0 + hf * 8 + r;
      int n = n0 + c * 16 + lc;
      int b = m >> 11, t = m & (kT - 1);
      int sec = n >> 10, cc = n & (kD - 1);
      int hh = cc >> 6, d = cc & 63;
      _Float16* dst = (sec == 0) ? qh : ((sec == 1) ? kh : vh);
      dst[(((size_t)(b * kH + hh) * kT + t) << 6) + d] = (_Float16)acc[c][r];
    }
  }
}

// --------------------------------------------------------------------------
// Flash-style causal attention. One wave per (b,h, 16-query tile).
// V tile (32 keys x 64, contiguous 4KB) is staged to LDS with async copies
// issued at loop top (overlaps QK^T + softmax); P round-trips LDS for the
// D-layout -> A-layout transpose.
// --------------------------------------------------------------------------
__global__ void attn_kernel(const _Float16* __restrict__ qh,
                            const _Float16* __restrict__ kh,
                            const _Float16* __restrict__ vh,
                            _Float16* __restrict__ yh) {
  __shared__ __align__(16) _Float16 Pst[4][16][32];   //  4 KB
  __shared__ __align__(16) _Float16 Vst[4][32][64];   // 16 KB
  const int lane = threadIdx.x & 31;
  const int wv   = threadIdx.x >> 5;
  const int hf   = lane >> 4;
  const int lc   = lane & 15;
  const int gw   = blockIdx.x * 4 + wv;
  const int qt   = gw & (kT / 16 - 1);
  const int bh   = gw >> 7;
  const int m0   = qt * 16;

  const _Float16* qb = qh + (size_t)bh * kT * kHD;
  const _Float16* kb = kh + (size_t)bh * kT * kHD;
  const _Float16* vb = vh + (size_t)bh * kT * kHD;

  // LDS byte offset of this wave's V staging buffer (low 32 bits of generic
  // pointer are the LDS offset per the aperture rules)
  const unsigned vstBase =
      (unsigned)(uintptr_t)(&Vst[wv][0][0]) + (unsigned)(lane * 16);

  const _Float16* qp = qb + (size_t)(m0 + lc) * kHD + hf * 16;
  v16h A0 = cat8(*(const v8h*)(qp),      *(const v8h*)(qp + 8));
  v16h A1 = cat8(*(const v8h*)(qp + 32), *(const v8h*)(qp + 40));

  float mrow[8], lrow[8];
#pragma unroll
  for (int r = 0; r < 8; ++r) { mrow[r] = -3.0e38f; lrow[r] = 0.0f; }
  v8f acc[4] = {};

  const int jend = m0 + 15;
  for (int j0 = 0; j0 <= jend; j0 += 32) {
    // Prior iteration's LDS reads must retire before the async engine
    // overwrites Vst (async LDS writes are unordered vs DS ops).
    asm volatile("s_wait_dscnt 0x0" ::: "memory");
    {
      const char* src = (const char*)(vb + (size_t)j0 * kHD) + lane * 16;
#pragma unroll
      for (int i = 0; i < 8; ++i) {
        unsigned ldsoff = vstBase + i * 512;
        const void* g = src + i * 512;
        asm volatile("global_load_async_to_lds_b128 %0, %1, off"
                     :: "v"(ldsoff), "v"(g) : "memory");
      }
    }

    // ---- S = Q K^T over this 32-key block (overlaps the async copy) ----
    v8f sLo = {}, sHi = {};
    {
      const _Float16* kp = kb + (size_t)(j0 + lc) * kHD + hf * 16;
      v16h B0 = cat8(*(const v8h*)(kp),      *(const v8h*)(kp + 8));
      v16h B1 = cat8(*(const v8h*)(kp + 32), *(const v8h*)(kp + 40));
      sLo = wmma_f16(A0, B0, sLo);
      sLo = wmma_f16(A1, B1, sLo);
    }
    {
      const _Float16* kp = kb + (size_t)(j0 + 16 + lc) * kHD + hf * 16;
      v16h B0 = cat8(*(const v8h*)(kp),      *(const v8h*)(kp + 8));
      v16h B1 = cat8(*(const v8h*)(kp + 32), *(const v8h*)(kp + 40));
      sHi = wmma_f16(A0, B0, sHi);
      sHi = wmma_f16(A1, B1, sHi);
    }

    // ---- online softmax ----
    const float scale = 0.125f;  // 1/sqrt(64)
#pragma unroll
    for (int r = 0; r < 8; ++r) {
      const int row = m0 + hf * 8 + r;
      float a = sLo[r] * scale;
      float b = sHi[r] * scale;
      if (j0 + lc      > row) a = -3.0e38f;
      if (j0 + 16 + lc > row) b = -3.0e38f;
      float rowM = redmax16(fmaxf(a, b));
      float mnew = fmaxf(mrow[r], rowM);
      float el = __expf(a - mnew);
      float eh = __expf(b - mnew);
      float corr = __expf(mrow[r] - mnew);
      float rowS = redsum16(el + eh);
      lrow[r] = lrow[r] * corr + rowS;
      mrow[r] = mnew;
      acc[0][r] *= corr; acc[1][r] *= corr;
      acc[2][r] *= corr; acc[3][r] *= corr;
      Pst[wv][hf * 8 + r][lc]      = (_Float16)el;
      Pst[wv][hf * 8 + r][16 + lc] = (_Float16)eh;
    }

    // D-layout -> A-layout transpose via LDS (same-wave, in-order DS pipe)
    asm volatile("s_wait_dscnt 0x0" ::: "memory");
    const _Float16* pp = &Pst[wv][lc][hf * 16];
    v16h PA = cat8(*(const v8h*)pp, *(const v8h*)(pp + 8));

    // ---- Y += P V, V fragments from the async-staged LDS tile ----
    asm volatile("s_wait_asynccnt 0x0" ::: "memory");
#pragma unroll
    for (int c = 0; c < 4; ++c) {
      v16h VB;
#pragma unroll
      for (int h2 = 0; h2 < 16; ++h2) {
        VB[h2] = Vst[wv][hf * 16 + h2][c * 16 + lc];
      }
      acc[c] = wmma_f16(PA, VB, acc[c]);
    }
  }

  const int b = bh >> 4, hh = bh & 15;
#pragma unroll
  for (int r = 0; r < 8; ++r) {
    float inv = 1.0f / lrow[r];
    size_t rowoff = (size_t)(b * kT + m0 + hf * 8 + r) * kD + hh * kHD;
#pragma unroll
    for (int c = 0; c < 4; ++c) {
      yh[rowoff + c * 16 + lc] = (_Float16)(acc[c][r] * inv);
    }
  }
}

// --------------------------------------------------------------------------
// out = y @ w_proj^T  (fp32 output). One wave per 16x64 tile, single-buffered.
// --------------------------------------------------------------------------
__global__ void proj_gemm(const _Float16* __restrict__ yhh,
                          const _Float16* __restrict__ wh,
                          float* __restrict__ out) {
  const int lane = threadIdx.x & 31;
  const int wv   = threadIdx.x >> 5;
  const int hf   = lane >> 4;
  const int lc   = lane & 15;
  const int gw   = blockIdx.x * 4 + wv;
  const int tilesN = kD / 64;  // 16
  const int m0 = (gw / tilesN) * 16;
  const int n0 = (gw % tilesN) * 64;

  v8f acc[4] = {};
  const _Float16* ap = yhh + (size_t)(m0 + lc) * kD + hf * 16;
  for (int kt = 0; kt < kD; kt += 32) {
    v16h A = cat8(*(const v8h*)(ap + kt), *(const v8h*)(ap + kt + 8));
    __builtin_prefetch(ap + kt + 64, 0, 3);
#pragma unroll
    for (int c = 0; c < 4; ++c) {
      const _Float16* bp = wh + (size_t)(n0 + c * 16 + lc) * kD + kt + hf * 16;
      v16h Bf = cat8(*(const v8h*)bp, *(const v8h*)(bp + 8));
      acc[c] = wmma_f16(A, Bf, acc[c]);
    }
  }
#pragma unroll
  for (int c = 0; c < 4; ++c) {
#pragma unroll
    for (int r = 0; r < 8; ++r) {
      int m = m0 + hf * 8 + r;
      int n = n0 + c * 16 + lc;
      out[(size_t)m * kD + n] = acc[c][r];
    }
  }
}

extern "C" void kernel_launch(void* const* d_in, const int* in_sizes, int n_in,
                              void* d_out, int out_size, void* d_ws, size_t ws_size,
                              hipStream_t stream) {
  (void)in_sizes; (void)n_in; (void)out_size; (void)ws_size;
  const float* x  = (const float*)d_in[0];   // [2,2048,1024]
  const float* wq = (const float*)d_in[1];   // [3072,1024]
  const float* wp = (const float*)d_in[2];   // [1024,1024]
  float* out = (float*)d_out;                // [2,2048,1024]

  char* ws = (char*)d_ws;
  size_t off = 0;
  _Float16* xh  = (_Float16*)(ws + off); off += (size_t)kM  * kD * 2;
  _Float16* wqh = (_Float16*)(ws + off); off += (size_t)kNQ * kD * 2;
  _Float16* wph = (_Float16*)(ws + off); off += (size_t)kD  * kD * 2;
  _Float16* qh  = (_Float16*)(ws + off); off += (size_t)kM  * kD * 2;
  _Float16* kh  = (_Float16*)(ws + off); off += (size_t)kM  * kD * 2;
  _Float16* vh  = (_Float16*)(ws + off); off += (size_t)kM  * kD * 2;
  _Float16* yh  = (_Float16*)(ws + off); off += (size_t)kM  * kD * 2;
  // total workspace: 48 MB

  cvt_f32_to_f16<<<2048, 256, 0, stream>>>(x,  xh,  kM * kD);
  cvt_f32_to_f16<<<2048, 256, 0, stream>>>(wq, wqh, kNQ * kD);
  cvt_f32_to_f16<<<1024, 256, 0, stream>>>(wp, wph, kD * kD);

  qkv_gemm<<<3072, 128, 0, stream>>>(xh, wqh, qh, kh, vh);
  attn_kernel<<<1024, 128, 0, stream>>>(qh, kh, vh, yh);
  proj_gemm<<<1024, 128, 0, stream>>>(yh, wph, out);
}
